// MelGaussianFilteredMSE_16406775071362
// MI455X (gfx1250) — matrix operand order repaired and checked by reading, not compiled, not run on hardware
//
#include <hip/hip_runtime.h>

// ---------------------------------------------------------------------------
// MelGaussianFilteredMSE for gfx1250 (MI455X, wave32, WMMA)
//
// mean( (M @ (10^(mo/10) - 10^(tg/10)))^2 )  with M banded (half-width <= 10)
//
// Band-blocked bf16 WMMA GEMM (16x16x32), f32 accum, double final reduce.
// K-tiles fixed at 3 and T padded to 2048 so the GEMM wave body is fully
// straight-line: no EXEC manipulation, address math folds into IOFFSETs.
// ---------------------------------------------------------------------------

typedef __bf16 v8bf  __attribute__((ext_vector_type(8)));
typedef __bf16 v16bf __attribute__((ext_vector_type(16)));
typedef float  v8f   __attribute__((ext_vector_type(8)));

static constexpr int BATCH = 32;
static constexpr int F     = 1025;   // frame size (rows of M, and K dim)
static constexpr int T     = 2000;
static constexpr int FPAD  = 1040;   // 65 * 16
static constexpr int KPAD  = 1056;   // 33 * 32
static constexpr int TPAD  = 2048;   // 128 * 16 (zero-padded tail tiles)
static constexpr int FT    = FPAD / 16;   // 65 f-tiles
static constexpr int KT_MAX = KPAD / 32 - 3;  // 30: highest legal kt_lo
static constexpr double INV_COUNT = 1.0 / (double(BATCH) * double(F) * double(T));
static constexpr float  LOG2_10_OVER_10 = 0.33219280948873623f; // log2(10)/10

// ---------------------------------------------------------------------------
// Fragment loaders (wave32); A and B use the same contiguous k order, k = 16h + s.
// A (16x32 bf16, row-major, ld elems): lane l -> row l&15;
//   lanes 0-15 hold K=0..15 contiguous, lanes 16-31 hold K=16..31.
// B (32x16 bf16 from K-contiguous N-major storage): lane l -> col l&15;
//   lanes 0-15 hold K=0..15 contiguous, lanes 16-31 hold K=16..31.
// ---------------------------------------------------------------------------
__device__ __forceinline__ v16bf load_a_frag(const __bf16* tile, int ld, int lane) {
    const int row = lane & 15;
    const int kb  = (lane >> 4) << 4;           // 0 or 16
    const __bf16* p = tile + (size_t)row * ld + kb;
    v8bf lo = *(const v8bf*)(p);                // K = kb .. kb+7      (16B load)
    v8bf hi = *(const v8bf*)(p + 8);            // K = kb+8 .. kb+15   (16B load)
    return __builtin_shufflevector(lo, hi, 0, 1, 2, 3, 4, 5, 6, 7,
                                           8, 9, 10, 11, 12, 13, 14, 15);
}

__device__ __forceinline__ v16bf load_b_frag(const __bf16* p) {
    return *(const v16bf*)p;                    // 16 contiguous K (32B load)
}

// ---------------------------------------------------------------------------
// Kernel 0: zero the double accumulator (ws is poisoned 0xAA by harness)
// ---------------------------------------------------------------------------
__global__ void k_init(double* accum) { *accum = 0.0; }

// ---------------------------------------------------------------------------
// Kernel 1: transform matrix f32 [1025][1025] -> bf16 [FPAD][KPAD], zero pad
// ---------------------------------------------------------------------------
__global__ __launch_bounds__(256)
void k_cvt_matrix(const float* __restrict__ M, __bf16* __restrict__ Mb) {
    int idx = blockIdx.x * 256 + threadIdx.x;
    if (idx >= FPAD * KPAD) return;
    int f = idx / KPAD;
    int k = idx - f * KPAD;
    float v = (f < F && k < F) ? M[(size_t)f * F + k] : 0.0f;
    Mb[idx] = (__bf16)v;
}

// ---------------------------------------------------------------------------
// Kernel 2: Dt[b][t][k] = 10^(mo[b][k][t]/10) - 10^(tg[b][k][t]/10)  (bf16)
// LDS-tiled 32x32 transpose so GEMM B-fragments are K-contiguous.
// Writes the full [TPAD][KPAD] slab (zeros in both pads) so the GEMM needs
// no bounds checks at all.
// Grid: (33 k-tiles, 64 t-tiles, 32 batches), block 256 = 32x8
// ---------------------------------------------------------------------------
__global__ __launch_bounds__(256)
void k_build_diff(const float* __restrict__ mo, const float* __restrict__ tg,
                  __bf16* __restrict__ Dt) {
    __shared__ __bf16 tile[32][33];             // +1 pad: bank-conflict free
    const int b  = blockIdx.z;
    const int k0 = blockIdx.x * 32;             // source f (== GEMM k)
    const int t0 = blockIdx.y * 32;
    const int tx = threadIdx.x & 31;
    const int ty = threadIdx.x >> 5;            // 0..7
    const size_t in_base = (size_t)b * F * T;
#pragma unroll
    for (int i = 0; i < 4; ++i) {
        int f = k0 + ty + 8 * i;
        int t = t0 + tx;
        float v = 0.0f;
        if (f < F && t < T) {
            size_t idx = in_base + (size_t)f * T + t;
            float a = exp2f(mo[idx] * LOG2_10_OVER_10);   // v_exp_f32
            float c = exp2f(tg[idx] * LOG2_10_OVER_10);
            v = a - c;
        }
        tile[ty + 8 * i][tx] = (__bf16)v;       // 0 in k-pad and t-pad
    }
    __syncthreads();
    __bf16* out = Dt + (size_t)b * TPAD * KPAD;
#pragma unroll
    for (int i = 0; i < 4; ++i) {
        int t = t0 + ty + 8 * i;                // always < TPAD (64*32)
        int k = k0 + tx;                        // always < KPAD (33*32)
        out[(size_t)t * KPAD + k] = tile[tx][ty + 8 * i];
    }
}

// ---------------------------------------------------------------------------
// Kernel 3: band-blocked WMMA GEMM + fused square-reduce.
// Grid: (65 f-tiles, 4 t-groups, 32 batches), block = 256 (8 waves).
// Each wave: one 16-row F tile x 4 T tiles; exactly 3 K-tiles of 32.
// Band of rows [f0,f0+16) lies in cols [f0-10, f0+25]; the 96-wide aligned
// window starting at clamp((f0-12)>>5, 0, 30)*32 always covers it, and the
// zero padding in Mb/Dt makes surplus tiles contribute exactly zero.
// No divergent branches anywhere -> EXEC stays all-ones (WMMA requirement),
// and all per-tile addressing folds into 24-bit instruction offsets.
// ---------------------------------------------------------------------------
__global__ __launch_bounds__(256)
void k_band_gemm(const __bf16* __restrict__ Mb, const __bf16* __restrict__ Dt,
                 double* __restrict__ accum) {
    const int lane = threadIdx.x & 31;
    const int wave = threadIdx.x >> 5;          // 0..7
    const int f0   = blockIdx.x * 16;           // 0..1024
    const int b    = blockIdx.z;

    int kt_lo = (f0 > 12 ? (f0 - 12) : 0) >> 5;
    if (kt_lo > KT_MAX) kt_lo = KT_MAX;         // keep 3 tiles inside KPAD

    // Load (and reuse) the 3 A fragments for this F tile -- static regs.
    const __bf16* Mtile = Mb + (size_t)f0 * KPAD + (size_t)kt_lo * 32;
    const v16bf a0 = load_a_frag(Mtile,      KPAD, lane);
    const v16bf a1 = load_a_frag(Mtile + 32, KPAD, lane);
    const v16bf a2 = load_a_frag(Mtile + 64, KPAD, lane);

    // Per-wave B base: lane -> column row; all tile/k offsets are constants.
    const int tt0 = blockIdx.y * 32 + wave * 4; // 4 t-tiles per wave, no tail
    const int col = lane & 15;
    const int kb  = (lane >> 4) << 4;           // 0 or 16
    const __bf16* Dwave = Dt
        + ((size_t)b * TPAD + (size_t)tt0 * 16 + col) * KPAD
        + (size_t)kt_lo * 32 + kb;

    float lsum = 0.0f;
#pragma unroll
    for (int j = 0; j < 4; ++j) {
        const __bf16* p = Dwave + (size_t)(j * 16) * KPAD;  // imm offset
        v8f c = {0.f, 0.f, 0.f, 0.f, 0.f, 0.f, 0.f, 0.f};
        c = __builtin_amdgcn_wmma_f32_16x16x32_bf16(
                false, a0, false, load_b_frag(p),      (short)0, c, false, false);
        c = __builtin_amdgcn_wmma_f32_16x16x32_bf16(
                false, a1, false, load_b_frag(p + 32), (short)0, c, false, false);
        c = __builtin_amdgcn_wmma_f32_16x16x32_bf16(
                false, a2, false, load_b_frag(p + 64), (short)0, c, false, false);
#pragma unroll
        for (int e = 0; e < 8; ++e) lsum += c[e] * c[e];
    }

    // wave32 butterfly reduce, then one f64 atomic per wave.
#pragma unroll
    for (int off = 16; off > 0; off >>= 1)
        lsum += __shfl_xor(lsum, off, 32);
    if (lane == 0) atomicAdd(accum, (double)lsum);
}

// ---------------------------------------------------------------------------
// Kernel 4: scalar finalize
// ---------------------------------------------------------------------------
__global__ void k_final(const double* __restrict__ accum, float* __restrict__ out) {
    out[0] = (float)(accum[0] * INV_COUNT);
}

// ---------------------------------------------------------------------------
extern "C" void kernel_launch(void* const* d_in, const int* in_sizes, int n_in,
                              void* d_out, int out_size, void* d_ws, size_t ws_size,
                              hipStream_t stream) {
    const float* mo = (const float*)d_in[0];          // (32, 1025, 2000) f32
    const float* tg = (const float*)d_in[1];          // (32, 1025, 2000) f32
    const float* M  = (const float*)d_in[2];          // (1025, 1025)     f32
    float* out = (float*)d_out;

    // Workspace layout
    char* ws = (char*)d_ws;
    double* accum = (double*)ws;                                  // 8 B
    __bf16* Mb = (__bf16*)(ws + 256);                             // 2.2 MB
    __bf16* Dt = (__bf16*)(ws + (size_t)4 * 1024 * 1024);         // ~132 MB

    k_init<<<1, 1, 0, stream>>>(accum);

    {
        int n = FPAD * KPAD;
        k_cvt_matrix<<<(n + 255) / 256, 256, 0, stream>>>(M, Mb);
    }

    {
        dim3 grid(KPAD / 32, TPAD / 32, BATCH);                   // (33, 64, 32)
        k_build_diff<<<grid, 256, 0, stream>>>(mo, tg, Dt);
    }

    {
        dim3 grid(FT, TPAD / 16 / 32, BATCH);                     // (65, 4, 32)
        k_band_gemm<<<grid, 256, 0, stream>>>(Mb, Dt, accum);
    }

    k_final<<<1, 1, 0, stream>>>(accum, out);
}
